// ContextAwareAttention_50130858278922
// MI455X (gfx1250) — matrix operand order repaired and checked by reading, not compiled, not run on hardware
//
#include <hip/hip_runtime.h>

// ---------------------------------------------------------------------------
// ContextAwareAttention for MI455X (gfx1250, wave32, WMMA bf16 16x16x32 + TDM)
// L=512, B=32, d=768, H=8, hd=96, NUM_REGIONS=196
// ---------------------------------------------------------------------------

typedef __attribute__((ext_vector_type(16))) __bf16          v16bf;
typedef __attribute__((ext_vector_type(8)))  float           v8f;
typedef __attribute__((ext_vector_type(16))) unsigned short  us16;
typedef __attribute__((ext_vector_type(4)))  unsigned short  us4;
typedef __attribute__((ext_vector_type(4)))  unsigned int    u32x4;
typedef __attribute__((ext_vector_type(8)))  int             i32x8;
typedef __attribute__((ext_vector_type(4)))  int             i32x4;

#define CA_L   512
#define CA_B   32
#define CA_D   768
#define CA_H   8
#define CA_HD  96
#define CA_NR  196
#define CA_T   (CA_L - CA_NR)   // 316
#define CA_M   (CA_L * CA_B)    // 16384 rows in flattened (L,B) order
#define CA_K   CA_D             // GEMM reduction dim

#if __has_include(<hip/amd_detail/amd_gfx1250_TDM.h>)
#define CA_TDM_ARGS6 1          // therock-10.0 headers -> 6-arg TDM builtin
#endif

__device__ __forceinline__ unsigned short f2bf(float f) {
  unsigned int u = __builtin_bit_cast(unsigned int, f);
  u += 0x7FFFu + ((u >> 16) & 1u);          // round-to-nearest-even
  return (unsigned short)(u >> 16);
}

__device__ __forceinline__ v8f wmma_bf16(v16bf a, v16bf b, v8f c) {
  // D = A(16x32) * B(32x16) + C, f32 accumulate
  return __builtin_amdgcn_wmma_f32_16x16x32_bf16(
      false, a, false, b, (short)0, c, false, false);
}

// A fragment (16x32, 16-bit layout): lane = row M (lane&15),
// elements 0..15 -> K = kh..kh+15,
// kh = 16*(lane>>4).  Source bf16 row-major, leading dim ldk.
__device__ __forceinline__ v16bf load_afrag(const unsigned short* A, int ldk,
                                            int m0, int kk, int lane) {
  const unsigned short* p =
      A + (size_t)(m0 + (lane & 15)) * ldk + kk + ((lane >> 4) << 4);
  us16 u;
#pragma unroll
  for (int i = 0; i < 16; ++i) u[i] = p[i];
  return __builtin_bit_cast(v16bf, u);
}

// B fragment (32x16, 16-bit layout): lane = column N (lane&15),
// elements 0..15 -> K = 16*(lane>>4)+0..15 (16 contiguous bf16 in source).
__device__ __forceinline__ v16bf load_bfrag(const unsigned short* p) {
  us16 u;
#pragma unroll
  for (int i = 0; i < 16; ++i) u[i] = p[i];
  return __builtin_bit_cast(v16bf, u);
}

// ---------------------------------------------------------------------------
// Tensor Data Mover: 2-D bf16 tile DMA global -> LDS (D# per ISA 08 §8).
// tile fully in-bounds: tensor dims == tile dims; dim0 contiguous in memory.
// ---------------------------------------------------------------------------
__device__ __forceinline__ void tdm_load_2d(unsigned lds_off,
                                            const unsigned short* gaddr,
                                            unsigned tile_d0, unsigned tile_d1,
                                            unsigned stride0_elems) {
  const unsigned long long ga = (unsigned long long)(const void*)gaddr;
  u32x4 g0;
  g0[0] = 1u;                                                 // count=1 (user D#)
  g0[1] = lds_off;                                            // LDS byte address
  g0[2] = (unsigned)ga;                                       // global addr [31:0]
  g0[3] = (unsigned)((ga >> 32) & 0x01FFFFFFu) | (2u << 30);  // [56:32] | type=2
  i32x8 g1;
  g1[0] = (int)(1u << 16);                     // wg_mask=0, data_size=1 -> 2 bytes
  g1[1] = (int)((tile_d0 & 0xFFFFu) << 16);    // tensor_dim0[15:0]
  g1[2] = (int)((tile_d1 & 0xFFFFu) << 16);    // tensor_dim0 hi=0 | tensor_dim1[15:0]
  g1[3] = (int)((tile_d0 & 0xFFFFu) << 16);    // tensor_dim1 hi=0 | tile_dim0
  g1[4] = (int)(tile_d1 & 0xFFFFu);            // tile_dim1 | tile_dim2=0
  g1[5] = (int)stride0_elems;                  // tensor_dim0_stride [31:0]
  g1[6] = 0;
  g1[7] = 0;
  const i32x4 z4 = {0, 0, 0, 0};
#ifdef CA_TDM_ARGS6
  const i32x8 z8 = {0, 0, 0, 0, 0, 0, 0, 0};
  __builtin_amdgcn_tensor_load_to_lds(g0, g1, z4, z4, z8, 0);
#else
  __builtin_amdgcn_tensor_load_to_lds(g0, g1, z4, z4, 0);
#endif
}

__device__ __forceinline__ unsigned lds_offset_of(const void* p) {
  return (unsigned)(unsigned long long)p;  // generic LDS ptr: offset in [31:0]
}

// ---------------------------------------------------------------------------
// fp32 -> bf16 bulk cast (one-time per tensor; removes per-GEMM convert VALU)
// ---------------------------------------------------------------------------
__global__ __launch_bounds__(256) void cast_bf16_k(
    const float* __restrict__ src, unsigned short* __restrict__ dst, int n) {
  const int i = (blockIdx.x * 256 + threadIdx.x) * 4;
  if (i + 3 < n) {
    const float4 v = *(const float4*)(src + i);
    us4 o;
    o[0] = f2bf(v.x); o[1] = f2bf(v.y); o[2] = f2bf(v.z); o[3] = f2bf(v.w);
    *(us4*)(dst + i) = o;
  }
}

// ---------------------------------------------------------------------------
// GEMM: C(M x 768) = A(M x 768) * W(768 x 768)^T, bf16 WMMA, f32 accumulate.
// ---------------------------------------------------------------------------
enum { OUT_QK = 0, OUT_VT = 1, OUT_ACC = 2 };

template <int MODE>
__device__ __forceinline__ void write_ctile(void* Cout, int mb, int nb, v8f c,
                                            float scale,
                                            const float* __restrict__ coefp,
                                            int accumulate, int lane) {
  const int mo = (lane >> 4) << 3;   // lanes 0-15 -> M=r, lanes 16-31 -> M=8+r
  const int n  = nb + (lane & 15);
  if constexpr (MODE == OUT_ACC) {
    const float coef = *coefp;
    float* O = (float*)Cout;
#pragma unroll
    for (int r = 0; r < 8; ++r) {
      const size_t idx = (size_t)(mb + mo + r) * CA_D + n;   // (L,B,d) flat
      const float v = coef * c[r];
      O[idx] = accumulate ? (O[idx] + v) : v;
    }
  } else {
    unsigned short* O = (unsigned short*)Cout;
    const int h = n / CA_HD, dd = n % CA_HD;
#pragma unroll
    for (int r = 0; r < 8; ++r) {
      const int m = mb + mo + r;
      const int l = m >> 5, b = m & 31;          // m = l*B + b
      const size_t idx =
          (MODE == OUT_QK)
              ? ((size_t)((b * CA_H + h) * CA_L + l) * CA_HD + dd)  // (B,H,L,hd)
              : ((size_t)((b * CA_H + h) * CA_HD + dd) * CA_L + l); // (B,H,hd,L)
      O[idx] = f2bf(scale * c[r]);
    }
  }
}

template <int MODE>
__global__ __launch_bounds__(128) void gemm_wmma_k(
    const unsigned short* __restrict__ A, const unsigned short* __restrict__ W,
    void* __restrict__ Cout, float scale, const float* __restrict__ coefp,
    int accumulate) {
  const int lane = threadIdx.x & 31;
  const int wave = threadIdx.x >> 5;
  const int m0 = blockIdx.y * 64 + (wave >> 1) * 32;
  const int n0 = blockIdx.x * 64 + (wave & 1) * 32;

  v8f c00 = {}, c01 = {}, c10 = {}, c11 = {};
  for (int kk = 0; kk < CA_K; kk += 32) {
    if (kk + 32 < CA_K) {  // stream hint for the next K slab of the weights
      __builtin_prefetch(W + (size_t)(n0 + (lane & 15)) * CA_K + kk + 32, 0, 1);
    }
    const v16bf a0 = load_afrag(A, CA_K, m0, kk, lane);
    const v16bf a1 = load_afrag(A, CA_K, m0 + 16, kk, lane);
    const int kb = kk + ((lane >> 4) << 4);
    const v16bf b0 = load_bfrag(W + (size_t)(n0 + (lane & 15)) * CA_K + kb);
    const v16bf b1 = load_bfrag(W + (size_t)(n0 + 16 + (lane & 15)) * CA_K + kb);
    c00 = wmma_bf16(a0, b0, c00);
    c01 = wmma_bf16(a0, b1, c01);
    c10 = wmma_bf16(a1, b0, c10);
    c11 = wmma_bf16(a1, b1, c11);
  }
  write_ctile<MODE>(Cout, m0,      n0,      c00, scale, coefp, accumulate, lane);
  write_ctile<MODE>(Cout, m0,      n0 + 16, c01, scale, coefp, accumulate, lane);
  write_ctile<MODE>(Cout, m0 + 16, n0,      c10, scale, coefp, accumulate, lane);
  write_ctile<MODE>(Cout, m0 + 16, n0 + 16, c11, scale, coefp, accumulate, lane);
}

// ---------------------------------------------------------------------------
// Flash attention: one wave per (b, h, 16-row q tile).
// K/V chunks staged in LDS via TDM (double buffered, TENSORcnt pipelined).
// Q,K in (B,H,L,hd) bf16 (Q pre-scaled by 1/sqrt(hd)); V in (B,H,hd,L) bf16.
// ---------------------------------------------------------------------------
__global__ __launch_bounds__(32) void attn_wmma_k(
    const unsigned short* __restrict__ Qp, const unsigned short* __restrict__ Kp,
    const unsigned short* __restrict__ Vt, unsigned short* __restrict__ Obf) {
  const int lane = threadIdx.x;
  const int bh = blockIdx.y, b = bh >> 3, h = bh & 7;
  const int qt = blockIdx.x * 16;
  const unsigned short* Qb = Qp + (size_t)bh * CA_L * CA_HD;
  const unsigned short* Kb = Kp + (size_t)bh * CA_L * CA_HD;
  const unsigned short* Vb = Vt + (size_t)bh * CA_HD * CA_L;

  __shared__ alignas(16) unsigned short bufK[2][32 * CA_HD];  // chunk rows x hd
  __shared__ alignas(16) unsigned short bufV[2][CA_HD * 32];  // hd rows x chunk
  __shared__ alignas(16) unsigned short pls[16 * 32];         // P C->A staging

  // Q A-fragments: 16 x 96 = three 16x32 fragments (registers for whole pass)
  v16bf qa[3];
  {
    const int row = lane & 15, kh = (lane >> 4) << 4;
#pragma unroll
    for (int t = 0; t < 3; ++t) {
      const unsigned short* p = Qb + (size_t)(qt + row) * CA_HD + t * 32 + kh;
      us16 u;
#pragma unroll
      for (int i = 0; i < 16; ++i) u[i] = p[i];
      qa[t] = __builtin_bit_cast(v16bf, u);
    }
  }

  v8f o[6];
  float mrow[8], srow[8];
#pragma unroll
  for (int nn = 0; nn < 6; ++nn) o[nn] = {};
#pragma unroll
  for (int r = 0; r < 8; ++r) { mrow[r] = -1.0e30f; srow[r] = 0.0f; }

  // prime the pipeline: chunk 0 into buffer 0 (TDM issues once per wave)
  tdm_load_2d(lds_offset_of(&bufK[0][0]), Kb, CA_HD, 32, CA_HD);
  tdm_load_2d(lds_offset_of(&bufV[0][0]), Vb, 32, CA_HD, CA_L);

  int buf = 0;
  for (int kc = 0; kc < CA_L; kc += 32) {
    const bool more = (kc + 32) < CA_L;
    if (more) {  // prefetch next chunk into the other buffer
      tdm_load_2d(lds_offset_of(&bufK[buf ^ 1][0]),
                  Kb + (size_t)(kc + 32) * CA_HD, CA_HD, 32, CA_HD);
      tdm_load_2d(lds_offset_of(&bufV[buf ^ 1][0]),
                  Vb + (kc + 32), 32, CA_HD, CA_L);
      __builtin_amdgcn_s_wait_tensorcnt(2);  // current buffer's 2 DMAs done
    } else {
      __builtin_amdgcn_s_wait_tensorcnt(0);
    }

    // ---- scores: two 16x16 tiles over this 32-key chunk (K frags from LDS)
    v8f s[2];
#pragma unroll
    for (int j = 0; j < 2; ++j) {
      v8f acc = {};
      const int krow = j * 16 + (lane & 15);          // chunk-local key row
      const int kb = (lane >> 4) << 4;
#pragma unroll
      for (int t = 0; t < 3; ++t) {
        acc = wmma_bf16(qa[t],
                        load_bfrag(&bufK[buf][krow * CA_HD + t * 32 + kb]),
                        acc);
      }
      s[j] = acc;
    }

    // ---- online softmax (row stats replicated across each 16-lane half) ---
    float fr[8];
#pragma unroll
    for (int r = 0; r < 8; ++r) {
      float mc = fmaxf(s[0][r], s[1][r]);
#pragma unroll
      for (int d = 1; d < 16; d <<= 1) mc = fmaxf(mc, __shfl_xor(mc, d, 32));
      const float mn = fmaxf(mrow[r], mc);
      fr[r] = __expf(mrow[r] - mn);
      mrow[r] = mn;
      const float p0 = __expf(s[0][r] - mn);
      const float p1 = __expf(s[1][r] - mn);
      s[0][r] = p0;
      s[1][r] = p1;
      float rs = p0 + p1;
#pragma unroll
      for (int d = 1; d < 16; d <<= 1) rs += __shfl_xor(rs, d, 32);
      srow[r] = srow[r] * fr[r] + rs;
    }
#pragma unroll
    for (int nn = 0; nn < 6; ++nn)
#pragma unroll
      for (int r = 0; r < 8; ++r) o[nn][r] *= fr[r];

    // ---- P: C-layout -> A-fragment layout via LDS -------------------------
    __syncthreads();  // previous iteration's fragment reads are done
    {
      const int mo = (lane >> 4) << 3, col = lane & 15;
#pragma unroll
      for (int j = 0; j < 2; ++j)
#pragma unroll
        for (int r = 0; r < 8; ++r)
          pls[(mo + r) * 32 + j * 16 + col] = f2bf(s[j][r]);
    }
    __syncthreads();
    v16bf pa;
    {
      const int row = lane & 15, kh = (lane >> 4) << 4;
      const unsigned short* p = &pls[row * 32 + kh];
      us16 u;
#pragma unroll
      for (int i = 0; i < 16; ++i) u[i] = p[i];
      pa = __builtin_bit_cast(v16bf, u);
    }

    // ---- P @ V : V tile in LDS is (hd rows x 32 chunk cols) ---------------
    {
      const int kb = (lane >> 4) << 4;
#pragma unroll
      for (int nn = 0; nn < 6; ++nn) {
        const int dd = nn * 16 + (lane & 15);
        o[nn] = wmma_bf16(pa, load_bfrag(&bufV[buf][dd * 32 + kb]), o[nn]);
      }
    }
    buf ^= 1;
  }

  // ---- finalize: divide by softmax sum, emit bf16 (L*B, d) ---------------
  const int mo = (lane >> 4) << 3, nlo = lane & 15;
#pragma unroll
  for (int r = 0; r < 8; ++r) {
    const float inv = 1.0f / srow[r];
    const int l = qt + mo + r;
    const size_t rowbase = ((size_t)l * CA_B + b) * CA_D + h * CA_HD;
#pragma unroll
    for (int nn = 0; nn < 6; ++nn)
      Obf[rowbase + nn * 16 + nlo] = f2bf(o[nn][r] * inv);
  }
}

// ---------------------------------------------------------------------------
// Window aggregation (fp32 math, bf16 output). One wave per (l, b).
// ---------------------------------------------------------------------------
template <int W, int SHIFT>
__global__ __launch_bounds__(32) void window_k(const float* __restrict__ q,
                                               unsigned short* __restrict__ out) {
  constexpr int PER = CA_D / 32;  // 24 channels per lane
  const int lane = threadIdx.x;
  const int l = blockIdx.x, b = blockIdx.y;
  unsigned short* orow = out + ((size_t)l * CA_B + b) * CA_D;

  if (l < CA_NR) {  // visual region: straight copy
    const float* src = q + ((size_t)l * CA_B + b) * CA_D;
#pragma unroll
    for (int i = 0; i < PER; ++i) orow[lane + 32 * i] = f2bf(src[lane + 32 * i]);
    return;
  }
  const int t = l - CA_NR;

  float a[PER];
  {
    const int at = t + SHIFT;
    if (at < CA_T) {
      const float* ar = q + ((size_t)(at + CA_NR) * CA_B + b) * CA_D;
#pragma unroll
      for (int i = 0; i < PER; ++i) a[i] = ar[lane + 32 * i];
    } else {
#pragma unroll
      for (int i = 0; i < PER; ++i) a[i] = 0.0f;
    }
  }
  float na2 = 0.0f;
#pragma unroll
  for (int i = 0; i < PER; ++i) na2 += a[i] * a[i];
#pragma unroll
  for (int d = 1; d < 32; d <<= 1) na2 += __shfl_xor(na2, d, 32);
  const float na = fmaxf(sqrtf(na2), 1e-8f);

  float o[PER];
#pragma unroll
  for (int i = 0; i < PER; ++i) o[i] = 0.0f;

#pragma unroll
  for (int w = 0; w < W; ++w) {
    const int kidx = t + w - W / 2;
    if (kidx < 0 || kidx >= CA_T) continue;
    const float* wr = q + ((size_t)(kidx + CA_NR) * CA_B + b) * CA_D;
    float wv[PER];
#pragma unroll
    for (int i = 0; i < PER; ++i) wv[i] = wr[lane + 32 * i];
    float dot = 0.0f, nw2 = 0.0f;
#pragma unroll
    for (int i = 0; i < PER; ++i) { dot += a[i] * wv[i]; nw2 += wv[i] * wv[i]; }
#pragma unroll
    for (int d = 1; d < 32; d <<= 1) {
      dot += __shfl_xor(dot, d, 32);
      nw2 += __shfl_xor(nw2, d, 32);
    }
    const float nw = fmaxf(sqrtf(nw2), 1e-8f);
    const float wgt = dot / (na * nw);
#pragma unroll
    for (int i = 0; i < PER; ++i) o[i] += wgt * wv[i];
  }
#pragma unroll
  for (int i = 0; i < PER; ++i) orow[lane + 32 * i] = f2bf(o[i]);
}

// ---------------------------------------------------------------------------
// Host launch
// ---------------------------------------------------------------------------
extern "C" void kernel_launch(void* const* d_in, const int* in_sizes, int n_in,
                              void* d_out, int out_size, void* d_ws,
                              size_t ws_size, hipStream_t stream) {
  (void)in_sizes; (void)n_in; (void)out_size; (void)ws_size;

  const float* query = (const float*)d_in[0];
  const float* key   = (const float*)d_in[1];
  const float* value = (const float*)d_in[2];
  const float* w_in [3] = {(const float*)d_in[3], (const float*)d_in[5],
                           (const float*)d_in[7]};
  const float* w_out[3] = {(const float*)d_in[4], (const float*)d_in[6],
                           (const float*)d_in[8]};
  const float* coef [3] = {(const float*)d_in[9], (const float*)d_in[10],
                           (const float*)d_in[11]};
  float* out = (float*)d_out;

  // workspace carve-up
  char* ws = (char*)d_ws;
  const size_t NB16 = (size_t)CA_M * CA_D * sizeof(unsigned short);  // 25.2 MB
  unsigned short* q2b  = (unsigned short*)ws; ws += NB16;  // query / windowed q (bf16)
  unsigned short* Kb16 = (unsigned short*)ws; ws += NB16;  // key (bf16)
  unsigned short* Vb16 = (unsigned short*)ws; ws += NB16;  // value (bf16)
  unsigned short* Qp   = (unsigned short*)ws; ws += NB16;  // (B,H,L,hd)
  unsigned short* Kp   = (unsigned short*)ws; ws += NB16;  // (B,H,L,hd)
  unsigned short* Vt   = (unsigned short*)ws; ws += NB16;  // (B,H,hd,L)
  unsigned short* Obf  = (unsigned short*)ws; ws += NB16;  // (L*B, d)
  unsigned short* Wb   = (unsigned short*)ws;              // per-stream weights
  const size_t DD = (size_t)CA_D * CA_D;
  ws += 4 * DD * sizeof(unsigned short);                   // 4.7 MB

  const dim3 gGemm(CA_D / 64, CA_M / 64);   // (12, 256), 128 threads
  const dim3 gAttn(CA_L / 16, CA_B * CA_H); // (32, 256), 32 threads
  const dim3 gWin(CA_L, CA_B);              // (512, 32), 32 threads
  const float qscale = 0.10206207261596577f;  // 1/sqrt(96)
  const int NE = CA_M * CA_D;               // 12.58M elems

  // one-time bf16 casts of the big activations
  cast_bf16_k<<<NE / 1024, 256, 0, stream>>>(query, q2b, NE);
  cast_bf16_k<<<NE / 1024, 256, 0, stream>>>(key, Kb16, NE);
  cast_bf16_k<<<NE / 1024, 256, 0, stream>>>(value, Vb16, NE);

  for (int s = 0; s < 3; ++s) {
    // per-stream weight casts (w_in: 3d x d, w_out: d x d)
    cast_bf16_k<<<(int)(3 * DD / 1024), 256, 0, stream>>>(w_in[s], Wb,
                                                          (int)(3 * DD));
    cast_bf16_k<<<(int)(DD / 1024), 256, 0, stream>>>(w_out[s], Wb + 3 * DD,
                                                      (int)DD);

    // Q-projection input: cast query for stream 0, windowed bf16 otherwise
    if (s == 1) window_k<3, 0><<<gWin, 32, 0, stream>>>(query, q2b);
    if (s == 2) window_k<5, 1><<<gWin, 32, 0, stream>>>(query, q2b);

    gemm_wmma_k<OUT_QK><<<gGemm, 128, 0, stream>>>(
        q2b, Wb, Qp, qscale, nullptr, 0);
    gemm_wmma_k<OUT_QK><<<gGemm, 128, 0, stream>>>(
        Kb16, Wb + DD, Kp, 1.0f, nullptr, 0);
    gemm_wmma_k<OUT_VT><<<gGemm, 128, 0, stream>>>(
        Vb16, Wb + 2 * DD, Vt, 1.0f, nullptr, 0);

    attn_wmma_k<<<gAttn, 32, 0, stream>>>(Qp, Kp, Vt, Obf);

    gemm_wmma_k<OUT_ACC><<<gGemm, 128, 0, stream>>>(
        Obf, Wb + 3 * DD, out, 1.0f, coef[s], s > 0);
  }
}
